// GraphConstruct_74285754351628
// MI455X (gfx1250) — matrix operand order repaired and checked by reading, not compiled, not run on hardware
//
#include <hip/hip_runtime.h>
#include <math.h>

// ---------------------------------------------------------------------------
// KNN graph construction for n=m=8192, e=256, K=16, SCALE=2 on gfx1250.
//
// Pipeline:
//  1) prep_kernel : split f32 inputs into 3 bf16 terms (hi/mid/lo) so the
//     distance GEMM can run on v_wmma_f32_16x16x32_bf16 with ~f32 accuracy
//     (6 cross products per K-chunk, error ~1e-6 rel); also row norms.
//  2) topk_kernel : per wave, 16 ye-rows; loop n-tiles of 16 xe-rows,
//     48 WMMAs per tile (8 K-chunks x 6 split products) accumulate y.x in
//     f32; distance = ||x||^2 - 2 y.x (||y||^2 is rank-invariant, added at
//     the end). LDS-transpose the 16x16 tile, two lanes per row keep running
//     sorted top-16 lists (threshold reject), merge halves at the end.
//     Writes score_out (exp(-d/10), broadcast over SCALE) and idx.
//  3) diff_kernel : stage 32 ye rows + 32 gathered xe rows in padded LDS,
//     emit |ye - xe[idx]| with m-contiguous coalesced stores, dup over SCALE.
// ---------------------------------------------------------------------------

#define N_PTS 8192
#define M_PTS 8192
#define E_DIM 256
#define KNN   16
#define SC    2

typedef __attribute__((ext_vector_type(16))) __bf16 v16bf;
typedef __attribute__((ext_vector_type(8)))  __bf16 v8bf;
typedef __attribute__((ext_vector_type(8)))  float  v8f;

#define WMMA_BF16(A, B, C) \
  __builtin_amdgcn_wmma_f32_16x16x32_bf16(false, (A), false, (B), (short)0, (C), false, false)

// ------------------------- Kernel 1: split + norms -------------------------
__global__ void __launch_bounds__(256) prep_kernel(
    const float* __restrict__ xe, const float* __restrict__ ye,
    __bf16* __restrict__ x0, __bf16* __restrict__ x1, __bf16* __restrict__ x2,
    __bf16* __restrict__ y0, __bf16* __restrict__ y1, __bf16* __restrict__ y2,
    float* __restrict__ xnorm, float* __restrict__ ynorm)
{
  __shared__ float red[256];
  const int row = blockIdx.x;          // 0..16383 : xe rows then ye rows
  const int t   = threadIdx.x;         // 0..255 = e index
  const float* src; __bf16 *d0, *d1, *d2; float* nrm; int r;
  if (row < N_PTS) { src = xe; d0 = x0; d1 = x1; d2 = x2; nrm = xnorm; r = row; }
  else             { src = ye; d0 = y0; d1 = y1; d2 = y2; nrm = ynorm; r = row - N_PTS; }

  const size_t o  = (size_t)r * E_DIM + t;
  const float  v  = src[o];
  const __bf16 h0 = (__bf16)v;
  const float  r1 = v - (float)h0;
  const __bf16 h1 = (__bf16)r1;
  const float  r2 = r1 - (float)h1;
  const __bf16 h2 = (__bf16)r2;
  d0[o] = h0; d1[o] = h1; d2[o] = h2;

  red[t] = v * v;
  __syncthreads();
  for (int s = 128; s > 0; s >>= 1) { if (t < s) red[t] += red[t + s]; __syncthreads(); }
  if (t == 0) nrm[r] = red[0];
}

// A-operand (16x32 bf16): lane holds row (lane&15); K = base..base+7 and
// base+16..base+23 with base = 8*(lane>>4). Two 16B loads, fused.
__device__ __forceinline__ v16bf load_a(const __bf16* p) {
  v8bf lo = *(const v8bf*)(p);
  v8bf hi = *(const v8bf*)(p + 16);
  return __builtin_shufflevector(lo, hi, 0,1,2,3,4,5,6,7,8,9,10,11,12,13,14,15);
}

// ------------------------- Kernel 2: WMMA + top-k --------------------------
__global__ void __launch_bounds__(128) topk_kernel(
    const __bf16* __restrict__ x0, const __bf16* __restrict__ x1, const __bf16* __restrict__ x2,
    const __bf16* __restrict__ y0, const __bf16* __restrict__ y1, const __bf16* __restrict__ y2,
    const float* __restrict__ xnorm, const float* __restrict__ ynorm,
    float* __restrict__ out_score, float* __restrict__ out_idxf, int* __restrict__ ws_idx)
{
  __shared__ float ldsD[4][16 * 20];     // per-wave 16x16 dist tile, stride 20
  __shared__ float mgd[4][32][16];       // per-wave merge scratch
  __shared__ int   mgi[4][32][16];

  const int lane = threadIdx.x & 31;
  const int wv   = threadIdx.x >> 5;     // 4 waves / block
  const int hi   = lane >> 4;            // K-half selector
  const int l15  = lane & 15;
  const int m0   = blockIdx.x * 64 + wv * 16;   // 16 ye rows per wave

  // A pointers (ye splits): lane-fixed across the whole n loop -> L1 resident
  const size_t aoff = (size_t)(m0 + l15) * E_DIM + hi * 8;
  const __bf16* pa0 = y0 + aoff;
  const __bf16* pa1 = y1 + aoff;
  const __bf16* pa2 = y2 + aoff;

  float bd[KNN]; int bi[KNN];            // sorted ascending, bd[15] = threshold
#pragma unroll
  for (int q = 0; q < KNN; ++q) { bd[q] = 3.0e38f; bi[q] = 0; }

  for (int n0 = 0; n0 < N_PTS; n0 += 16) {
    // B-operand (32x16 = xe^T tile): lane holds column (lane&15),
    // K = base..base+7 and base+16..base+23 with base = 8*(lane>>4) (same as A).
    const size_t boff = (size_t)(n0 + l15) * E_DIM + hi * 8;
    const __bf16* pb0 = x0 + boff;
    const __bf16* pb1 = x1 + boff;
    const __bf16* pb2 = x2 + boff;

    v8f c = {0.f, 0.f, 0.f, 0.f, 0.f, 0.f, 0.f, 0.f};
#pragma unroll
    for (int kc = 0; kc < 8; ++kc) {
      const int k0 = kc * 32;
      v16bf a0 = load_a(pa0 + k0);
      v16bf a1 = load_a(pa1 + k0);
      v16bf a2 = load_a(pa2 + k0);
      v16bf b0 = load_a(pb0 + k0);
      v16bf b1 = load_a(pb1 + k0);
      v16bf b2 = load_a(pb2 + k0);
      // 3-term split: keep products down to 2^-17 relative (f32-grade dot)
      c = WMMA_BF16(a0, b0, c);
      c = WMMA_BF16(a0, b1, c);
      c = WMMA_BF16(a1, b0, c);
      c = WMMA_BF16(a1, b1, c);
      c = WMMA_BF16(a0, b2, c);
      c = WMMA_BF16(a2, b0, c);
    }

    // selection key: ||x_n||^2 - 2 y.x  (||y_m||^2 constant per row)
    const float xn = xnorm[n0 + l15];
#pragma unroll
    for (int j = 0; j < 8; ++j)          // C: lane holds N=l15, M=j+8*hi
      ldsD[wv][(j + 8 * hi) * 20 + l15] = fmaf(-2.0f, c[j], xn);
    __builtin_amdgcn_wave_barrier();     // intra-wave LDS ops are in-order

    // lanes r and r+16 scan row r, columns hi*8..hi*8+7
#pragma unroll
    for (int cc = 0; cc < 8; ++cc) {
      const float d = ldsD[wv][l15 * 20 + hi * 8 + cc];
      if (d < bd[KNN - 1]) {             // rare after warm-up (~100 inserts/row)
        const int n = n0 + hi * 8 + cc;
        int p = KNN - 1;
#pragma unroll
        for (int q = KNN - 1; q >= 1; --q) {
          const bool mv = bd[q - 1] > d; // strict: ties keep earlier (lower n)
          if (mv) { bd[q] = bd[q - 1]; bi[q] = bi[q - 1]; p = q - 1; }
        }
        bd[p] = d; bi[p] = n;
      }
    }
    __builtin_amdgcn_wave_barrier();
  }

  // merge the two half-lists of each row
#pragma unroll
  for (int q = 0; q < KNN; ++q) { mgd[wv][lane][q] = bd[q]; mgi[wv][lane][q] = bi[q]; }
  __builtin_amdgcn_wave_barrier();

  if (lane < 16) {
    const int m = m0 + lane;
    const float yn = ynorm[m];
    int ia = 0, ib = 0;
#pragma unroll
    for (int t = 0; t < KNN; ++t) {
      const float da = mgd[wv][lane][ia];      const int na = mgi[wv][lane][ia];
      const float db = mgd[wv][lane + 16][ib]; const int nb = mgi[wv][lane + 16][ib];
      const bool ta = (da < db) || (da == db && na < nb);  // ties -> lower idx
      const float dsel = ta ? da : db;
      const int   nsel = ta ? na : nb;
      ia += ta ? 1 : 0; ib += ta ? 0 : 1;
      const float dfull = dsel + yn;           // full squared distance
      const float sc = expf(dfull * (-0.1f));
      out_score[(size_t)(t * SC + 0) * M_PTS + m] = sc;    // [K][SCALE][m]
      out_score[(size_t)(t * SC + 1) * M_PTS + m] = sc;
      out_idxf[(size_t)m * KNN + t] = (float)nsel;         // [m][K] as float
      ws_idx[(size_t)m * KNN + t]   = nsel;
    }
  }
}

// ------------------------- Kernel 3: diff_patch ----------------------------
__global__ void __launch_bounds__(256) diff_kernel(
    const float* __restrict__ xe, const float* __restrict__ ye,
    const int* __restrict__ ws_idx, float* __restrict__ out_diff)
{
  __shared__ float sy[32][257];   // stride 257 -> conflict-free transpose read
  __shared__ float sx[32][257];
  const int t  = threadIdx.x;     // 0..255
  const int m0 = blockIdx.x * 32; // 32 ye rows per block
  const int k  = blockIdx.y;      // neighbor slot

#pragma unroll 4
  for (int i = 0; i < 32; ++i) {
    sy[i][t] = ye[(size_t)(m0 + i) * E_DIM + t];           // coalesced rows
    const int xr = ws_idx[(size_t)(m0 + i) * KNN + k];     // uniform -> scalar
    sx[i][t] = xe[(size_t)xr * E_DIM + t];                 // coalesced gather
  }
  __syncthreads();

  const int j  = t & 31;          // m within tile (contiguous across lanes)
  const int eb = t >> 5;          // 0..7
#pragma unroll 4
  for (int eo = 0; eo < 32; ++eo) {
    const int e = eo * 8 + eb;
    const float d = fabsf(sy[j][e] - sx[j][e]);
    const size_t base = ((size_t)(k * E_DIM + e) * SC) * M_PTS + m0 + j;
    out_diff[base] = d;           // SCALE broadcast: s=0 and s=1
    out_diff[base + M_PTS] = d;
  }
}

// ------------------------------ launcher -----------------------------------
extern "C" void kernel_launch(void* const* d_in, const int* in_sizes, int n_in,
                              void* d_out, int out_size, void* d_ws, size_t ws_size,
                              hipStream_t stream)
{
  const float* xe = (const float*)d_in[0];   // [8192, 256] f32
  const float* ye = (const float*)d_in[1];   // [8192, 256] f32

  char* ws = (char*)d_ws;                    // ~25.1 MB used
  const size_t SPL = (size_t)N_PTS * E_DIM * sizeof(__bf16);   // 4 MB
  __bf16* x0 = (__bf16*)(ws + 0 * SPL);
  __bf16* x1 = (__bf16*)(ws + 1 * SPL);
  __bf16* x2 = (__bf16*)(ws + 2 * SPL);
  __bf16* y0 = (__bf16*)(ws + 3 * SPL);
  __bf16* y1 = (__bf16*)(ws + 4 * SPL);
  __bf16* y2 = (__bf16*)(ws + 5 * SPL);
  float* xnorm = (float*)(ws + 6 * SPL);
  float* ynorm = xnorm + N_PTS;
  int*   ws_idx = (int*)(ynorm + M_PTS);

  // outputs concatenated flat: score [16][2*8192], idx [8192][16] (as float),
  // diff_patch [16*256][2*8192]
  float* out_score = (float*)d_out;
  float* out_idxf  = out_score + (size_t)KNN * SC * M_PTS;   // +262144
  float* out_diff  = out_idxf + (size_t)M_PTS * KNN;         // +131072

  prep_kernel<<<N_PTS + M_PTS, 256, 0, stream>>>(xe, ye, x0, x1, x2, y0, y1, y2,
                                                 xnorm, ynorm);
  topk_kernel<<<M_PTS / 64, 128, 0, stream>>>(x0, x1, x2, y0, y1, y2,
                                              xnorm, ynorm,
                                              out_score, out_idxf, ws_idx);
  diff_kernel<<<dim3(M_PTS / 32, KNN), 256, 0, stream>>>(xe, ye, ws_idx, out_diff);
}
